// Attention_51548197487430
// MI455X (gfx1250) — matrix operand order repaired and checked by reading, not compiled, not run on hardware
//
#include <hip/hip_runtime.h>
#include <hip/hip_bf16.h>

typedef __attribute__((ext_vector_type(16))) _Float16 v16h;
typedef __attribute__((ext_vector_type(8)))  _Float16 v8h;
typedef __attribute__((ext_vector_type(8)))  float    v8f;

// clang's async-LDS builtin takes generic (gcc-style) int4 pointers:
//   __builtin_amdgcn_global_load_async_to_lds_b128(V4i*1, V4i*3, Ii, Ii)
typedef int v4i __attribute__((vector_size(16)));
typedef __attribute__((address_space(1))) v4i* as1_v4i;
typedef __attribute__((address_space(3))) v4i* as3_v4i;

#define B_   2
#define S_   2048
#define D_   1024
#define H_   16
#define HD_  64
#define SCALE_ 0.125f   // 64^-0.5

#define HAS_ASYNC_LDS __has_builtin(__builtin_amdgcn_global_load_async_to_lds_b128)

// ---------------------------------------------------------------------------
// helpers
// ---------------------------------------------------------------------------
__device__ __forceinline__ v8f wmma16(v16h a, v16h b, v8f c) {
  return __builtin_amdgcn_wmma_f32_16x16x32_f16(false, a, false, b,
                                                (short)0, c, false, false);
}

// A-fragment (16x32 f16): lane m=l&15 holds K = 2*kbase+{0..15},
// kbase = (l>>4)*8.  Two contiguous 16B chunks per lane.
__device__ __forceinline__ v16h load_a(const _Float16* rowp, int kbase) {
  v8h lo = *(const v8h*)(rowp + 2 * kbase);
  v8h hi = *(const v8h*)(rowp + 2 * kbase + 8);
  return __builtin_shufflevector(lo, hi, 0,1,2,3,4,5,6,7,8,9,10,11,12,13,14,15);
}

// 16B global -> LDS copy; async DMA path on CDNA5, sync fallback otherwise
__device__ __forceinline__ void cp16_g2l(void* l, const void* g) {
#if HAS_ASYNC_LDS
  __builtin_amdgcn_global_load_async_to_lds_b128((as1_v4i)(g), (as3_v4i)(l), 0, 0);
#else
  *(v8h*)l = *(const v8h*)g;
#endif
}

template <int N>
__device__ __forceinline__ void async_wait_le() {
#if HAS_ASYNC_LDS
  __builtin_amdgcn_s_wait_asynccnt(N);
#endif
}

__device__ __forceinline__ float redmax16(float v) {
  v = fmaxf(v, __shfl_xor(v, 1, 32));
  v = fmaxf(v, __shfl_xor(v, 2, 32));
  v = fmaxf(v, __shfl_xor(v, 4, 32));
  v = fmaxf(v, __shfl_xor(v, 8, 32));
  return v;
}
__device__ __forceinline__ float redsum16(float v) {
  v += __shfl_xor(v, 1, 32);
  v += __shfl_xor(v, 2, 32);
  v += __shfl_xor(v, 4, 32);
  v += __shfl_xor(v, 8, 32);
  return v;
}

// ---------------------------------------------------------------------------
// Kernel 0: fp32 -> f16 convert
// ---------------------------------------------------------------------------
__global__ void cvt_f32_f16_k(const float* __restrict__ src,
                              _Float16* __restrict__ dst, int n) {
  int i = blockIdx.x * blockDim.x + threadIdx.x;
  if (i < n) dst[i] = (_Float16)src[i];
}

// ---------------------------------------------------------------------------
// GEMM body shared by kernels 1 and 3: wave tile 64M x 32N, K stepped by 64
// with explicit ping-pong fragment sets (no register-rotation copies).
// ---------------------------------------------------------------------------
#define GEMM_LOOP(AR, BR0, BR1, ACC)                                         \
  v16h a0[4], a1[4], b0[2], b1[2];                                           \
  _Pragma("unroll")                                                          \
  for (int i = 0; i < 4; ++i) a0[i] = load_a(AR[i], kbase);                  \
  b0[0] = *(const v16h*)(BR0);                                               \
  b0[1] = *(const v16h*)(BR1);                                               \
  for (int kk = 0; kk < D_; kk += 64) {                                      \
    const int k1 = kk + 32; /* always < D_ */                                \
    __builtin_prefetch(AR[0] + kk + 256, 0, 1);                              \
    _Pragma("unroll")                                                        \
    for (int i = 0; i < 4; ++i) a1[i] = load_a(AR[i] + k1, kbase);           \
    b1[0] = *(const v16h*)(BR0 + k1);                                        \
    b1[1] = *(const v16h*)(BR1 + k1);                                        \
    _Pragma("unroll")                                                        \
    for (int i = 0; i < 4; ++i) {                                            \
      ACC[i][0] = wmma16(a0[i], b0[0], ACC[i][0]);                           \
      ACC[i][1] = wmma16(a0[i], b0[1], ACC[i][1]);                           \
    }                                                                        \
    const int k2 = (kk + 64 < D_) ? kk + 64 : kk; /* last: harmless reload */\
    _Pragma("unroll")                                                        \
    for (int i = 0; i < 4; ++i) a0[i] = load_a(AR[i] + k2, kbase);           \
    b0[0] = *(const v16h*)(BR0 + k2);                                        \
    b0[1] = *(const v16h*)(BR1 + k2);                                        \
    _Pragma("unroll")                                                        \
    for (int i = 0; i < 4; ++i) {                                            \
      ACC[i][0] = wmma16(a1[i], b1[0], ACC[i][0]);                           \
      ACC[i][1] = wmma16(a1[i], b1[1], ACC[i][1]);                           \
    }                                                                        \
  }

// ---------------------------------------------------------------------------
// Kernel 1: QKV projection.  M=4096, N=3072, K=1024.
// Block = 8 waves, tile 128M x 128N; wave tile 64M x 32N.
// Outputs: qf,kf f16 [B][H][S][HD]; v transposed vT f16 [B][H][HD][S].
// ---------------------------------------------------------------------------
__global__ __launch_bounds__(256) void qkv_gemm_k(
    const _Float16* __restrict__ xf, const _Float16* __restrict__ wf,
    _Float16* __restrict__ qf, _Float16* __restrict__ kf,
    _Float16* __restrict__ vT) {
  const int lane = threadIdx.x & 31, wave = threadIdx.x >> 5;
  const int wm = wave & 1, wn = wave >> 1;
  const int row0 = blockIdx.x * 128 + wm * 64;
  const int col0 = blockIdx.y * 128 + wn * 32;
  const int m = lane & 15, grp = lane >> 4, kbase = grp * 8;

  const _Float16* ar[4];
#pragma unroll
  for (int i = 0; i < 4; ++i) ar[i] = xf + (size_t)(row0 + i * 16 + m) * D_;
  const _Float16* br0 = wf + (size_t)(col0 + m) * D_ + grp * 16;
  const _Float16* br1 = wf + (size_t)(col0 + 16 + m) * D_ + grp * 16;

  v8f acc[4][2] = {};
  GEMM_LOOP(ar, br0, br1, acc)

  const int rbase = grp * 8;
#pragma unroll
  for (int i = 0; i < 4; ++i) {
#pragma unroll
    for (int j = 0; j < 2; ++j) {
      const int c = col0 + j * 16 + m;
      const int e = c >> 10;                 // 0=v, 1=q, 2=k (reference order)
      const int rem = c & 1023;
      const int h = rem >> 6, hd = rem & 63;
#pragma unroll
      for (int r = 0; r < 8; ++r) {
        const int row = row0 + i * 16 + rbase + r;
        const int bb = row >> 11, s = row & 2047;
        const _Float16 val = (_Float16)acc[i][j][r];
        const size_t bh = (size_t)(bb * H_ + h);
        if (e == 0)      vT[(bh * HD_ + hd) * S_ + s] = val;
        else if (e == 1) qf[(bh * S_ + s) * HD_ + hd] = val;
        else             kf[(bh * S_ + s) * HD_ + hd] = val;
      }
    }
  }
}

// ---------------------------------------------------------------------------
// Kernel 2: flash attention.  grid = (B*H, S/128), block = 256 (8 waves).
// 64-key blocks; K/V tiles staged into LDS with async global->LDS copies,
// double buffered, shared by all 8 waves.  16 WMMAs per wave per block.
// ---------------------------------------------------------------------------
__global__ __launch_bounds__(256) void flash_attn_k(
    const _Float16* __restrict__ qf, const _Float16* __restrict__ kf,
    const _Float16* __restrict__ vT, _Float16* __restrict__ of) {
  __shared__ _Float16 ldsK[2][64 * 64];   // [kv 64][hd 64]
  __shared__ _Float16 ldsV[2][64 * 64];   // [hd 64][kv 64]
  __shared__ _Float16 ldsP[8][16 * 64];   // per-wave P tile (16 q x 64 kv)

  const int t = threadIdx.x;
  const int lane = t & 31, wave = t >> 5;
  const int bh = blockIdx.x, b = bh >> 4, h = bh & 15;
  const int qrow0 = blockIdx.y * 128 + wave * 16;
  const int m = lane & 15, grp = lane >> 4;
  const int kbase = grp * 8, rbase = grp * 8, n = m;

  const _Float16* qb = qf + ((size_t)bh * S_ + qrow0) * HD_;
  const _Float16* kb = kf + (size_t)bh * S_ * HD_;
  const _Float16* vb = vT + (size_t)bh * HD_ * S_;

  // cooperative staging: each 8KB tile = 512 x 16B chunks, 2 passes of 256
  const int srow = t >> 3, scol = (t & 7) * 8;   // 32 rows per pass

  // prologue: stage kv=0 into buffer 0 (4 async ops per thread)
#pragma unroll
  for (int i = 0; i < 2; ++i) {
    const int rw = srow + 32 * i;
    cp16_g2l(&ldsK[0][rw * 64 + scol], kb + (size_t)rw * HD_ + scol);
    cp16_g2l(&ldsV[0][rw * 64 + scol], vb + (size_t)rw * S_ + scol);
  }

  const v16h aq0 = load_a(qb + m * HD_, kbase);
  const v16h aq1 = load_a(qb + m * HD_ + 32, kbase);

  v8f o[4] = {};
  float mr[8], lr[8], alpha[8];
#pragma unroll
  for (int r = 0; r < 8; ++r) { mr[r] = -1e30f; lr[r] = 0.0f; }

  _Float16* P = ldsP[wave];

  for (int kv = 0; kv < S_; kv += 64) {
    const int cur = (kv >> 6) & 1, nxt = cur ^ 1;
    const int kvn = (kv + 64 < S_) ? kv + 64 : kv;

    // stage next block while computing this one
#pragma unroll
    for (int i = 0; i < 2; ++i) {
      const int rw = srow + 32 * i;
      cp16_g2l(&ldsK[nxt][rw * 64 + scol],
               kb + (size_t)(kvn + rw) * HD_ + scol);
      cp16_g2l(&ldsV[nxt][rw * 64 + scol],
               vb + (size_t)rw * S_ + kvn + scol);
    }
    // all but the 4 just-issued copies (in-order completion) => current done
    async_wait_le<4>();
    __syncthreads();

    const _Float16* Kt = ldsK[cur];
    const _Float16* Vt = ldsV[cur];

    // ---- scores: four 16x16 tiles over HD = 2 x 32
    v8f sc[4];
#pragma unroll
    for (int jj = 0; jj < 4; ++jj) {
      v8f s = {};
      v16h bk0 = *(const v16h*)(Kt + (jj * 16 + n) * 64 + grp * 16);
      v16h bk1 = *(const v16h*)(Kt + (jj * 16 + n) * 64 + 32 + grp * 16);
      s = wmma16(aq0, bk0, s);
      s = wmma16(aq1, bk1, s);
      sc[jj] = s;
    }

    // ---- online softmax (row r spans the 16 lanes of this half-wave)
#pragma unroll
    for (int r = 0; r < 8; ++r) {
      const float v0 = sc[0][r] * SCALE_;
      const float v1 = sc[1][r] * SCALE_;
      const float v2 = sc[2][r] * SCALE_;
      const float v3 = sc[3][r] * SCALE_;
      const float bm = redmax16(fmaxf(fmaxf(v0, v1), fmaxf(v2, v3)));
      const float mn = fmaxf(mr[r], bm);
      const float p0 = __expf(v0 - mn);
      const float p1 = __expf(v1 - mn);
      const float p2 = __expf(v2 - mn);
      const float p3 = __expf(v3 - mn);
      const float rs = redsum16((p0 + p1) + (p2 + p3));
      const float al = __expf(mr[r] - mn);
      lr[r] = lr[r] * al + rs;
      mr[r] = mn;
      alpha[r] = al;
      _Float16* pr = P + (rbase + r) * 64 + n;
      pr[0]  = (_Float16)p0;
      pr[16] = (_Float16)p1;
      pr[32] = (_Float16)p2;
      pr[48] = (_Float16)p3;
    }

#pragma unroll
    for (int j = 0; j < 4; ++j)
#pragma unroll
      for (int r = 0; r < 8; ++r) o[j][r] *= alpha[r];

    // same-wave DS ops are in-order; drain before re-reading P as A-fragment
    asm volatile("s_wait_dscnt 0" ::: "memory");
    v16h ap0 = load_a(P + m * 64, kbase);        // kv 0..31
    v16h ap1 = load_a(P + m * 64 + 32, kbase);   // kv 32..63

    // ---- o += P * V  (K-dim = 64 kv -> 2 WMMA steps per hd tile)
#pragma unroll
    for (int j = 0; j < 4; ++j) {
      v16h bv0 = *(const v16h*)(Vt + (j * 16 + n) * 64 + grp * 16);
      v16h bv1 = *(const v16h*)(Vt + (j * 16 + n) * 64 + 32 + grp * 16);
      o[j] = wmma16(ap0, bv0, o[j]);
      o[j] = wmma16(ap1, bv1, o[j]);
    }

    __syncthreads();   // all waves done reading cur before it is overwritten
  }

  // ---- finalize
#pragma unroll
  for (int r = 0; r < 8; ++r) alpha[r] = 1.0f / lr[r];
#pragma unroll
  for (int j = 0; j < 4; ++j) {
#pragma unroll
    for (int r = 0; r < 8; ++r) {
      const int s = qrow0 + rbase + r;
      of[((size_t)b * S_ + s) * D_ + h * HD_ + j * 16 + n] =
          (_Float16)(o[j][r] * alpha[r]);
    }
  }
}

// ---------------------------------------------------------------------------
// Kernel 3: out projection + bias.  M=4096, N=1024, K=1024.
// ---------------------------------------------------------------------------
__global__ __launch_bounds__(256) void out_proj_k(
    const _Float16* __restrict__ of, const _Float16* __restrict__ wf,
    const float* __restrict__ b_out, float* __restrict__ out) {
  const int lane = threadIdx.x & 31, wave = threadIdx.x >> 5;
  const int wm = wave & 1, wn = wave >> 1;
  const int row0 = blockIdx.x * 128 + wm * 64;
  const int col0 = blockIdx.y * 128 + wn * 32;
  const int m = lane & 15, grp = lane >> 4, kbase = grp * 8;

  const _Float16* ar[4];
#pragma unroll
  for (int i = 0; i < 4; ++i) ar[i] = of + (size_t)(row0 + i * 16 + m) * D_;
  const _Float16* br0 = wf + (size_t)(col0 + m) * D_ + grp * 16;
  const _Float16* br1 = wf + (size_t)(col0 + 16 + m) * D_ + grp * 16;

  v8f acc[4][2] = {};
  GEMM_LOOP(ar, br0, br1, acc)

  const int rbase = grp * 8;
#pragma unroll
  for (int j = 0; j < 2; ++j) {
    const int c = col0 + j * 16 + m;
    const float bias = b_out[c];
#pragma unroll
    for (int i = 0; i < 4; ++i) {
#pragma unroll
      for (int r = 0; r < 8; ++r) {
        const int row = row0 + i * 16 + rbase + r;
        out[(size_t)row * D_ + c] = acc[i][j][r] + bias;
      }
    }
  }
}

// ---------------------------------------------------------------------------
// launch
// ---------------------------------------------------------------------------
extern "C" void kernel_launch(void* const* d_in, const int* in_sizes, int n_in,
                              void* d_out, int out_size, void* d_ws, size_t ws_size,
                              hipStream_t stream) {
  const float* x     = (const float*)d_in[0];
  const float* w_qkv = (const float*)d_in[1];
  const float* w_out = (const float*)d_in[2];
  const float* b_out = (const float*)d_in[3];
  float* out = (float*)d_out;

  const size_t NX  = (size_t)B_ * S_ * D_;   // 4,194,304
  const size_t NWQ = (size_t)3 * D_ * D_;    // 3,145,728
  const size_t NWO = (size_t)D_ * D_;        // 1,048,576

  char* ws = (char*)d_ws;
  _Float16* xf    = (_Float16*)ws; ws += NX  * 2;
  _Float16* wqkvf = (_Float16*)ws; ws += NWQ * 2;
  _Float16* woutf = (_Float16*)ws; ws += NWO * 2;
  _Float16* qf    = (_Float16*)ws; ws += NX  * 2;
  _Float16* kf    = (_Float16*)ws; ws += NX  * 2;
  _Float16* vT    = (_Float16*)ws; ws += NX  * 2;
  _Float16* of    = (_Float16*)ws;           // total ws use: 48 MB

  cvt_f32_f16_k<<<(int)(NX  / 256), 256, 0, stream>>>(x,     xf,    (int)NX);
  cvt_f32_f16_k<<<(int)(NWQ / 256), 256, 0, stream>>>(w_qkv, wqkvf, (int)NWQ);
  cvt_f32_f16_k<<<(int)(NWO / 256), 256, 0, stream>>>(w_out, woutf, (int)NWO);

  qkv_gemm_k  <<<dim3(32, 24), 256, 0, stream>>>(xf, wqkvf, qf, kf, vT);
  flash_attn_k<<<dim3(B_ * H_, S_ / 128), 256, 0, stream>>>(qf, kf, vT, of);
  out_proj_k  <<<dim3(32, 8), 256, 0, stream>>>(of, woutf, b_out, out);
}
